// ViT_3539053052671
// MI455X (gfx1250) — compile-verified, matrix operand order checked by reading, not run on hardware
//
#include <hip/hip_runtime.h>
#include <cmath>

// ---------------- types ----------------
typedef __bf16 bf16;
typedef bf16  v16bf  __attribute__((ext_vector_type(16)));
typedef float v8f    __attribute__((ext_vector_type(8)));
typedef bf16  bf16x2 __attribute__((ext_vector_type(2)));
typedef unsigned int u32x4 __attribute__((ext_vector_type(4)));
typedef int  i32x4 __attribute__((ext_vector_type(4)));
typedef int  i32x8 __attribute__((ext_vector_type(8)));

// ---------------- problem constants ----------------
#define E_DIM   512
#define NHEAD   8
#define DHEAD   64
#define NLAYER  6
#define NTOK    1025
#define NBATCH  16
#define NPATCH  1024
#define PDIM    768     // CIN*P*P
#define HID     2048    // RM*E
#define NCLS    1000
#define QKVN    1536
#define MROWS   (NBATCH*NTOK)                // 16400
#define BHND    (NBATCH*NHEAD*NTOK*DHEAD)    // 8,396,800 elems per Q/K/V plane

// ---------------- helpers ----------------
__device__ inline float gelu_f(float x) {
  return 0.5f * x * (1.0f + erff(x * 0.70710678118654752f));
}

__device__ inline v8f wmma_bf16(v16bf a, v16bf b, v8f c) {
  return __builtin_amdgcn_wmma_f32_16x16x32_bf16(false, a, false, b,
                                                 (short)0, c, false, false);
}

// Load one 16x32 bf16 fragment (A-layout, or B from an [N][K] tile).
// 16 rows on (lane&15); K contiguous at p+row*stride. Per CDNA5 ISA the
// lane group g owns K offsets +8 inside each 16-K half; pairs are contiguous,
// so each half-fragment folds into one ds_load_b128.
__device__ inline v16bf load_frag(const bf16* p, int stride, int lane, int kofs) {
  v16bf f;
  int m = lane & 15, g = lane >> 4;
  const bf16* rp = p + m * stride + kofs;
#pragma unroll
  for (int r = 0; r < 8; ++r) {
    int kk = ((r >> 2) << 4) + (g << 3) + ((r & 3) << 1);
    bf16x2 pr = *(const bf16x2*)(rp + kk);
    f[2 * r]     = pr.x;
    f[2 * r + 1] = pr.y;
  }
  return f;
}

// ---------------- Tensor Data Mover: 2D bf16 tile -> LDS ----------------
// D# per CDNA5 ISA. data_size=2B. Tensor dims are the
// REMAINING extents from the tile start so the TDM zero-pads M/N edges for us.
__device__ inline void tdm_load_2d(unsigned lds_off, const void* gaddr,
                                   int rem_d0, int rem_d1,
                                   int tile_d0, int tile_d1, int stride_elems)
{
  unsigned long long ga = (unsigned long long)(uintptr_t)gaddr;
  u32x4 g0;
  g0[0] = 1u;                                           // count=1, user D#
  g0[1] = lds_off;                                      // lds_addr (bytes)
  g0[2] = (unsigned)(ga & 0xffffffffu);                 // global_addr[31:0]
  g0[3] = (unsigned)((ga >> 32) & 0x01ffffffu) | (2u << 30); // addr[56:32]|type=2
  unsigned d0 = (unsigned)rem_d0, d1 = (unsigned)rem_d1;
  unsigned st = (unsigned)stride_elems;
  i32x8 g1;
  g1[0] = (int)(1u << 16);                              // data_size=2B, mask=0
  g1[1] = (int)((d0 & 0xffffu) << 16);                  // tensor_dim0[15:0]
  g1[2] = (int)(((d0 >> 16) & 0xffffu) | ((d1 & 0xffffu) << 16));
  g1[3] = (int)(((d1 >> 16) & 0xffffu) | ((unsigned)tile_d0 << 16)); // tile_dim0
  g1[4] = (int)((unsigned)tile_d1 & 0xffffu);           // tile_dim1; tile_dim2=0
  g1[5] = (int)st;                                      // tensor_dim0_stride lo
  g1[6] = 0;                                            // stride hi / dim1_stride lo
  g1[7] = 0;
  i32x4 z4 = {0, 0, 0, 0};
#if __clang_major__ >= 23
  i32x8 z8 = {0, 0, 0, 0, 0, 0, 0, 0};
  __builtin_amdgcn_tensor_load_to_lds(g0, g1, z4, z4, z8, 0);
#else
  __builtin_amdgcn_tensor_load_to_lds(g0, g1, z4, z4, 0);
#endif
}

__device__ inline unsigned lds_off_u32(const void* p) {
  return (unsigned)(uintptr_t)p;   // LDS aperture: addr[31:0] is the LDS offset
}

// ---------------- generic WMMA GEMM (TDM-staged, double-buffered) ----------
// C[M,N] = epi( A[M,K]_bf16 @ Wt[N,K]_bf16^T + bias ... )
// MODE: 0=f32 out, 1=f32 out + residual, 2=bf16 out, 3=bf16 out + GELU,
//       4=qkv split (Q/K [bh][t][d], V transposed [bh][d][t])
template <int MODE>
__global__ __launch_bounds__(128) void gemm_wmma(
    const bf16* __restrict__ A, const bf16* __restrict__ Wt,
    const float* __restrict__ bias, const float* __restrict__ resid,
    float* __restrict__ Cf, bf16* __restrict__ Cb,
    int M, int N, int K)
{
  __shared__ alignas(16) bf16 As[2][64][32];
  __shared__ alignas(16) bf16 Ws[2][64][32];

  const int tid  = threadIdx.x;
  const int wave = tid >> 5, lane = tid & 31;
  const int m0 = blockIdx.y * 64, n0 = blockIdx.x * 64;
  const int mq = (wave >> 1) * 32, nq = (wave & 1) * 32;
  const int ksteps = K >> 5;

  if (wave == 0) {
    tdm_load_2d(lds_off_u32(&As[0][0][0]), A + (size_t)m0 * K, K, M - m0, 32, 64, K);
    tdm_load_2d(lds_off_u32(&Ws[0][0][0]), Wt + (size_t)n0 * K, K, N - n0, 32, 64, K);
  }

  v8f zacc;
#pragma unroll
  for (int i = 0; i < 8; ++i) zacc[i] = 0.f;
  v8f acc[2][2];
#pragma unroll
  for (int i = 0; i < 2; ++i)
#pragma unroll
    for (int j = 0; j < 2; ++j) acc[i][j] = zacc;

  for (int ks = 0; ks < ksteps; ++ks) {
    const int buf = ks & 1;
    if (wave == 0) {
      if (ks + 1 < ksteps) {
        int k1 = (ks + 1) << 5;
        tdm_load_2d(lds_off_u32(&As[buf ^ 1][0][0]), A + (size_t)m0 * K + k1,
                    K - k1, M - m0, 32, 64, K);
        tdm_load_2d(lds_off_u32(&Ws[buf ^ 1][0][0]), Wt + (size_t)n0 * K + k1,
                    K - k1, N - n0, 32, 64, K);
        __builtin_amdgcn_s_wait_tensorcnt(2);
      } else {
        __builtin_amdgcn_s_wait_tensorcnt(0);
      }
    }
    __syncthreads();               // current buffer landed

    v16bf af[2], bw[2];
#pragma unroll
    for (int mf = 0; mf < 2; ++mf)
      af[mf] = load_frag(&As[buf][mq + mf * 16][0], 32, lane, 0);
#pragma unroll
    for (int nf = 0; nf < 2; ++nf)
      bw[nf] = load_frag(&Ws[buf][nq + nf * 16][0], 32, lane, 0);

#pragma unroll
    for (int mf = 0; mf < 2; ++mf)
#pragma unroll
      for (int nf = 0; nf < 2; ++nf)
        acc[mf][nf] = wmma_bf16(af[mf], bw[nf], acc[mf][nf]);
    __syncthreads();               // done reading before TDM overwrites
  }

  // epilogue; C layout: VGPR v -> M = v + 8*(lane>>4); N = lane&15
  const int lh = lane & 15, g = lane >> 4;
#pragma unroll
  for (int mf = 0; mf < 2; ++mf)
#pragma unroll
    for (int nf = 0; nf < 2; ++nf)
#pragma unroll
      for (int v = 0; v < 8; ++v) {
        int row = m0 + mq + mf * 16 + g * 8 + v;
        int col = n0 + nq + nf * 16 + lh;
        if (row < M && col < N) {
          float val = acc[mf][nf][v] + bias[col];
          if (MODE == 3) val = gelu_f(val);
          if (MODE == 4) {
            int which = col % 3;
            int d = (col / 3) & (DHEAD - 1);
            int hh = col / (3 * DHEAD);
            int bb = row / NTOK, tt = row % NTOK;
            int bh = bb * NHEAD + hh;
            size_t dst;
            if (which < 2)
              dst = (size_t)which * BHND + ((size_t)bh * NTOK + tt) * DHEAD + d;
            else
              dst = (size_t)2 * BHND + ((size_t)bh * DHEAD + d) * NTOK + tt;
            Cb[dst] = (bf16)val;
          } else {
            size_t off = (size_t)row * N + col;
            if (MODE == 1) val += resid[off];
            if (MODE == 2 || MODE == 3) Cb[off] = (bf16)val;
            else                        Cf[off] = val;
          }
        }
      }
}

// ---------------- flash attention (causal, TDM-staged) ----------------
// qkv3: bf16, Q/K planes [bh][t][64], V plane transposed [bh][64][t].
// O: bf16 [B*NTOK, E]. 128 query rows per WG (32/wave), 32-key tiles,
// WMMA for S=QK^T and O=PV, online softmax, K/V double-buffered via TDM.
__global__ __launch_bounds__(128) void attn_kernel(
    const bf16* __restrict__ qkv3, bf16* __restrict__ O)
{
  __shared__ alignas(16) bf16 Qs[128][64];
  __shared__ alignas(16) bf16 Ks[2][32][64];
  __shared__ alignas(16) bf16 Vt[2][64][32];
  __shared__ alignas(16) bf16 Ps[4][32][32];

  const int tid = threadIdx.x, wave = tid >> 5, lane = tid & 31;
  const int r0 = blockIdx.x * 128;
  const int h = blockIdx.y, b = blockIdx.z;
  const int bh = b * NHEAD + h;
  const int lh = lane & 15, g = lane >> 4;
  const int wq = wave * 32;

  const bf16* Qg = qkv3;
  const bf16* Kg = qkv3 + (size_t)BHND;
  const bf16* Vg = qkv3 + (size_t)2 * BHND;

  const int jmax = min(r0 + 127, NTOK - 1);
  const int ntiles = (jmax >> 5) + 1;

  if (wave == 0) {
    tdm_load_2d(lds_off_u32(&Qs[0][0]), Qg + ((size_t)bh * NTOK + r0) * DHEAD,
                DHEAD, NTOK - r0, DHEAD, 128, DHEAD);
    tdm_load_2d(lds_off_u32(&Ks[0][0][0]), Kg + ((size_t)bh * NTOK) * DHEAD,
                DHEAD, NTOK, DHEAD, 32, DHEAD);
    tdm_load_2d(lds_off_u32(&Vt[0][0][0]), Vg + (size_t)bh * DHEAD * NTOK,
                NTOK, DHEAD, 32, DHEAD, NTOK);
  }

  v8f zacc;
#pragma unroll
  for (int i = 0; i < 8; ++i) zacc[i] = 0.f;
  v8f acc[2][4];
  float mstate[2][8], lstate[2][8];
#pragma unroll
  for (int mf = 0; mf < 2; ++mf) {
#pragma unroll
    for (int nf = 0; nf < 4; ++nf) acc[mf][nf] = zacc;
#pragma unroll
    for (int v = 0; v < 8; ++v) { mstate[mf][v] = -1e30f; lstate[mf][v] = 0.f; }
  }

  v16bf aq[2][2];

  for (int ti = 0; ti < ntiles; ++ti) {
    const int jt = ti << 5;
    const int buf = ti & 1;
    if (wave == 0) {
      if (ti + 1 < ntiles) {
        int j1 = (ti + 1) << 5;
        tdm_load_2d(lds_off_u32(&Ks[buf ^ 1][0][0]),
                    Kg + ((size_t)bh * NTOK + j1) * DHEAD,
                    DHEAD, NTOK - j1, DHEAD, 32, DHEAD);
        tdm_load_2d(lds_off_u32(&Vt[buf ^ 1][0][0]),
                    Vg + (size_t)bh * DHEAD * NTOK + j1,
                    NTOK - j1, DHEAD, 32, DHEAD, NTOK);
        __builtin_amdgcn_s_wait_tensorcnt(2);
      } else {
        __builtin_amdgcn_s_wait_tensorcnt(0);
      }
    }
    __syncthreads();               // Q (first iter) + current K/V landed

    if (ti == 0) {
#pragma unroll
      for (int mf = 0; mf < 2; ++mf)
#pragma unroll
        for (int ks = 0; ks < 2; ++ks)
          aq[mf][ks] = load_frag(&Qs[wq + mf * 16][0], 64, lane, ks * 32);
    }

    // S = Q @ K^T
    v8f s[2][2];
#pragma unroll
    for (int mf = 0; mf < 2; ++mf)
#pragma unroll
      for (int nf = 0; nf < 2; ++nf) {
        v8f a = zacc;
#pragma unroll
        for (int ks = 0; ks < 2; ++ks) {
          v16bf bk = load_frag(&Ks[buf][nf * 16][0], 64, lane, ks * 32);
          a = wmma_bf16(aq[mf][ks], bk, a);
        }
        s[mf][nf] = a;
      }

    // scale + causal mask
#pragma unroll
    for (int mf = 0; mf < 2; ++mf)
#pragma unroll
      for (int nf = 0; nf < 2; ++nf)
#pragma unroll
        for (int v = 0; v < 8; ++v) {
          int i_g = r0 + wq + mf * 16 + g * 8 + v;
          int j_g = jt + nf * 16 + lh;
          float sv = s[mf][nf][v] * 0.125f;   // DH^-0.5
          if (j_g > i_g) sv = -1e30f;
          s[mf][nf][v] = sv;
        }

    // online softmax: each row lives on a 16-lane group (xor 1,2,4,8)
#pragma unroll
    for (int mf = 0; mf < 2; ++mf)
#pragma unroll
      for (int v = 0; v < 8; ++v) {
        float cm = fmaxf(s[mf][0][v], s[mf][1][v]);
#pragma unroll
        for (int msk = 8; msk >= 1; msk >>= 1)
          cm = fmaxf(cm, __shfl_xor(cm, msk, 32));
        float mold = mstate[mf][v];
        float mnew = fmaxf(mold, cm);
        float scale = __expf(mold - mnew);
        float s0 = s[mf][0][v], s1 = s[mf][1][v];
        float p0 = (s0 > -1e29f) ? __expf(s0 - mnew) : 0.f;
        float p1 = (s1 > -1e29f) ? __expf(s1 - mnew) : 0.f;
        s[mf][0][v] = p0; s[mf][1][v] = p1;
        float rs = p0 + p1;
#pragma unroll
        for (int msk = 8; msk >= 1; msk >>= 1)
          rs += __shfl_xor(rs, msk, 32);
        lstate[mf][v] = lstate[mf][v] * scale + rs;
        mstate[mf][v] = mnew;
#pragma unroll
        for (int nf = 0; nf < 4; ++nf) acc[mf][nf][v] *= scale;
      }

    // P: C-layout -> row-major LDS (per-wave region) for the A side of P@V
#pragma unroll
    for (int mf = 0; mf < 2; ++mf)
#pragma unroll
      for (int nf = 0; nf < 2; ++nf)
#pragma unroll
        for (int v = 0; v < 8; ++v)
          Ps[wave][mf * 16 + g * 8 + v][nf * 16 + lh] = (bf16)s[mf][nf][v];
    __syncthreads();

    // O += P @ V
    v16bf ap[2];
#pragma unroll
    for (int mf = 0; mf < 2; ++mf)
      ap[mf] = load_frag(&Ps[wave][mf * 16][0], 32, lane, 0);
#pragma unroll
    for (int nf = 0; nf < 4; ++nf) {
      v16bf bv = load_frag(&Vt[buf][nf * 16][0], 32, lane, 0);
#pragma unroll
      for (int mf = 0; mf < 2; ++mf)
        acc[mf][nf] = wmma_bf16(ap[mf], bv, acc[mf][nf]);
    }
    __syncthreads();               // buffers free for next TDM tile
  }

  // normalize + store O[b, t, h*DH + d] (bf16)
#pragma unroll
  for (int mf = 0; mf < 2; ++mf)
#pragma unroll
    for (int nf = 0; nf < 4; ++nf)
#pragma unroll
      for (int v = 0; v < 8; ++v) {
        int t = r0 + wq + mf * 16 + g * 8 + v;
        if (t < NTOK) {
          float ov = acc[mf][nf][v] / lstate[mf][v];
          O[((size_t)b * NTOK + t) * E_DIM + h * DHEAD + nf * 16 + lh] = (bf16)ov;
        }
      }
}

// ---------------- LayerNorm (one wave per 512-wide row; bf16 out) --------
__global__ __launch_bounds__(256) void ln_kernel(
    const float* __restrict__ X, long long xstride,
    const float* __restrict__ gamma, const float* __restrict__ beta,
    bf16* __restrict__ Y, long long ystride, int rows)
{
  int wave = threadIdx.x >> 5, lane = threadIdx.x & 31;
  int row = blockIdx.x * 8 + wave;
  if (row >= rows) return;
  const float* x = X + (size_t)row * xstride;
  float vals[16];
  float s = 0.f;
#pragma unroll
  for (int i = 0; i < 16; ++i) { vals[i] = x[lane + i * 32]; s += vals[i]; }
#pragma unroll
  for (int m = 16; m >= 1; m >>= 1) s += __shfl_xor(s, m, 32);
  float mean = s * (1.f / 512.f);
  float vs = 0.f;
#pragma unroll
  for (int i = 0; i < 16; ++i) { float d = vals[i] - mean; vs += d * d; }
#pragma unroll
  for (int m = 16; m >= 1; m >>= 1) vs += __shfl_xor(vs, m, 32);
  float rstd = rsqrtf(vs * (1.f / 512.f) + 1e-5f);
  bf16* y = Y + (size_t)row * ystride;
#pragma unroll
  for (int i = 0; i < 16; ++i) {
    int c = lane + i * 32;
    y[c] = (bf16)((vals[i] - mean) * rstd * gamma[c] + beta[c]);
  }
}

// ---------------- weight convert: f32 [K,N] -> bf16 transposed [N,K] -----
__global__ void convw_kernel(const float* __restrict__ W, bf16* __restrict__ Wt,
                             int K, int N)
{
  int idx = blockIdx.x * 256 + threadIdx.x;
  if (idx >= K * N) return;
  int n = idx % N, k = idx / N;          // coalesced read along N
  Wt[(size_t)n * K + k] = (bf16)W[idx];
}

// ---------------- im2col: x[B,3,512,512] -> patches bf16 [B*1024, 768] ----
__global__ void im2col_kernel(const float* __restrict__ x, bf16* __restrict__ patches)
{
  int idx = blockIdx.x * 256 + threadIdx.x;
  const int total = NBATCH * NPATCH * PDIM;
  if (idx >= total) return;
  int cp = idx % PDIM;
  int t  = (idx / PDIM) % NPATCH;
  int b  = idx / (PDIM * NPATCH);
  int c = cp >> 8, py = (cp >> 4) & 15, px = cp & 15;
  int hp = t >> 5, wp = t & 31;
  float v = x[(((size_t)b * 3 + c) * 512 + hp * 16 + py) * 512 + wp * 16 + px];
  patches[idx] = (bf16)v;
}

// ---------------- assemble z = concat(cls, patch_emb) + pos_embed ---------
__global__ void assemble_z(const float* __restrict__ pe, const float* __restrict__ cls,
                           const float* __restrict__ pos, float* __restrict__ z)
{
  int idx = blockIdx.x * 256 + threadIdx.x;
  const int total = MROWS * E_DIM;
  if (idx >= total) return;
  int e = idx & 511, row = idx >> 9;
  int b = row / NTOK, t = row % NTOK;
  float v;
  if (t == 0) v = cls[e] + pos[e];
  else        v = pe[((size_t)(b * NPATCH + t - 1)) * E_DIM + e] + pos[(size_t)t * E_DIM + e];
  z[idx] = v;
}

// ---------------- host-side orchestration ----------------
extern "C" void kernel_launch(void* const* d_in, const int* in_sizes, int n_in,
                              void* d_out, int out_size, void* d_ws, size_t ws_size,
                              hipStream_t stream) {
  (void)in_sizes; (void)n_in; (void)out_size; (void)ws_size;

  const float* xin     = (const float*)d_in[0];
  const float* conv_w  = (const float*)d_in[1];
  const float* conv_b  = (const float*)d_in[2];
  const float* cls     = (const float*)d_in[3];
  const float* pos     = (const float*)d_in[4];
  const float* h_ln_g  = (const float*)d_in[5 + 12 * NLAYER + 0];
  const float* h_ln_b  = (const float*)d_in[5 + 12 * NLAYER + 1];
  const float* h_fc1_w = (const float*)d_in[5 + 12 * NLAYER + 2];
  const float* h_fc1_b = (const float*)d_in[5 + 12 * NLAYER + 3];
  const float* h_fc2_w = (const float*)d_in[5 + 12 * NLAYER + 4];
  const float* h_fc2_b = (const float*)d_in[5 + 12 * NLAYER + 5];
  float* out = (float*)d_out;

  // workspace layout (~154.9 MiB total):
  //   z    f32  [16400,512]                        @ 0
  //   R1   bf16 [16400,512]  (LN out y / attn O)   @ 33,587,200
  //   R2   67,174,400 B: patches bf16 + patch-emb f32 tmp (pre-layer),
  //        aliases qkv3 bf16 [3*BHND] and h1 bf16 [16400,2048]
  //   warena: bf16 transposed weights (44,728,320 B)
  //   yh/hh head temporaries
  char* wsb = (char*)d_ws;
  float* z      = (float*)wsb;
  bf16*  y      = (bf16*)(wsb + 33587200);
  bf16*  Obuf   = y;                            // disjoint lifetime vs y
  char*  R2     = wsb + 50380800;
  bf16*  patches= (bf16*)R2;
  float* tmpP   = (float*)(R2 + 25165824);
  bf16*  qkv3   = (bf16*)R2;
  bf16*  h1     = (bf16*)R2;
  bf16*  warena = (bf16*)(wsb + 117555200);
  bf16*  yh     = (bf16*)(wsb + 162283520);
  bf16*  hh     = (bf16*)(wsb + 162299904);

  // --- convert all weights to bf16, transposed [N][K] ---
  bf16* conv_wt = warena;
  size_t woff = (size_t)PDIM * E_DIM;
  bf16 *qkv_wt[NLAYER], *out_wt[NLAYER], *fc1_wt[NLAYER], *fc2_wt[NLAYER];
  for (int l = 0; l < NLAYER; ++l) {
    qkv_wt[l] = warena + woff; woff += (size_t)E_DIM * QKVN;
    out_wt[l] = warena + woff; woff += (size_t)E_DIM * E_DIM;
    fc1_wt[l] = warena + woff; woff += (size_t)E_DIM * HID;
    fc2_wt[l] = warena + woff; woff += (size_t)HID * E_DIM;
  }
  bf16* hfc1_wt = warena + woff; woff += (size_t)E_DIM * HID;
  bf16* hfc2_wt = warena + woff;

  auto convw = [&](const float* W, bf16* Wt, int K, int N) {
    convw_kernel<<<(K * N + 255) / 256, 256, 0, stream>>>(W, Wt, K, N);
  };
  convw(conv_w, conv_wt, PDIM, E_DIM);
  for (int l = 0; l < NLAYER; ++l) {
    convw((const float*)d_in[5 + 12 * l + 2],  qkv_wt[l], E_DIM, QKVN);
    convw((const float*)d_in[5 + 12 * l + 4],  out_wt[l], E_DIM, E_DIM);
    convw((const float*)d_in[5 + 12 * l + 8],  fc1_wt[l], E_DIM, HID);
    convw((const float*)d_in[5 + 12 * l + 10], fc2_wt[l], HID, E_DIM);
  }
  convw(h_fc1_w, hfc1_wt, E_DIM, HID);
  convw(h_fc2_w, hfc2_wt, HID, NCLS);

  // --- patch embedding ---
  im2col_kernel<<<(NBATCH * NPATCH * PDIM + 255) / 256, 256, 0, stream>>>(xin, patches);
  gemm_wmma<0><<<dim3(E_DIM / 64, (NBATCH * NPATCH + 63) / 64), 128, 0, stream>>>(
      patches, conv_wt, conv_b, nullptr, tmpP, nullptr,
      NBATCH * NPATCH, E_DIM, PDIM);
  assemble_z<<<(MROWS * E_DIM + 255) / 256, 256, 0, stream>>>(tmpP, cls, pos, z);

  const dim3 gN512((E_DIM + 63) / 64, (MROWS + 63) / 64);
  const dim3 gQKV((QKVN + 63) / 64, (MROWS + 63) / 64);
  const dim3 gHID((HID + 63) / 64, (MROWS + 63) / 64);
  const int lnBlocks = (MROWS + 7) / 8;

  for (int l = 0; l < NLAYER; ++l) {
    const float* ln1_g = (const float*)d_in[5 + 12 * l + 0];
    const float* ln1_b = (const float*)d_in[5 + 12 * l + 1];
    const float* qkv_b = (const float*)d_in[5 + 12 * l + 3];
    const float* out_b = (const float*)d_in[5 + 12 * l + 5];
    const float* ln2_g = (const float*)d_in[5 + 12 * l + 6];
    const float* ln2_b = (const float*)d_in[5 + 12 * l + 7];
    const float* fc1_b = (const float*)d_in[5 + 12 * l + 9];
    const float* fc2_b = (const float*)d_in[5 + 12 * l + 11];

    ln_kernel<<<lnBlocks, 256, 0, stream>>>(z, E_DIM, ln1_g, ln1_b, y, E_DIM, MROWS);
    gemm_wmma<4><<<gQKV, 128, 0, stream>>>(
        y, qkv_wt[l], qkv_b, nullptr, nullptr, qkv3, MROWS, QKVN, E_DIM);
    attn_kernel<<<dim3((NTOK + 127) / 128, NHEAD, NBATCH), 128, 0, stream>>>(qkv3, Obuf);
    gemm_wmma<1><<<gN512, 128, 0, stream>>>(
        Obuf, out_wt[l], out_b, z, z, nullptr, MROWS, E_DIM, E_DIM);
    ln_kernel<<<lnBlocks, 256, 0, stream>>>(z, E_DIM, ln2_g, ln2_b, y, E_DIM, MROWS);
    gemm_wmma<3><<<gHID, 128, 0, stream>>>(
        y, fc1_wt[l], fc1_b, nullptr, nullptr, h1, MROWS, HID, E_DIM);
    gemm_wmma<1><<<gN512, 128, 0, stream>>>(
        h1, fc2_wt[l], fc2_b, z, z, nullptr, MROWS, E_DIM, HID);
  }

  // head: pooled = z[:,0,:], LN, MLP -> out [16,1000] (f32)
  ln_kernel<<<2, 256, 0, stream>>>(z, (long long)NTOK * E_DIM, h_ln_g, h_ln_b,
                                   yh, E_DIM, NBATCH);
  gemm_wmma<3><<<dim3(HID / 64, 1), 128, 0, stream>>>(
      yh, hfc1_wt, h_fc1_b, nullptr, nullptr, hh, NBATCH, HID, E_DIM);
  gemm_wmma<0><<<dim3((NCLS + 63) / 64, 1), 128, 0, stream>>>(
      hh, hfc2_wt, h_fc2_b, nullptr, out, nullptr, NBATCH, NCLS, HID);
}
